// MultiHeadAttention_65025804861696
// MI455X (gfx1250) — matrix operand order repaired and checked by reading, not compiled, not run on hardware
//
#include <hip/hip_runtime.h>
#include <hip/hip_bf16.h>

typedef __attribute__((ext_vector_type(16))) __bf16 bf16x16;
typedef __attribute__((ext_vector_type(8)))  __bf16 bf16x8;
typedef __attribute__((ext_vector_type(8)))  float  f32x8;

constexpr int Bc = 4, Sc = 2048, Ec = 1024, Hc = 16, Dc = 64;
constexpr int Mc = Bc * Sc; // 8192

// ---------------------------------------------------------------------------
// Fragment helpers (CDNA5 WMMA 16x16x32 bf16 layouts, wave32)
// A 16x32: lane holds row M=lane%16; elements 0..7 -> K = half*8 + i,
//          elements 8..15 -> K = half*8 + 16 + i   (half = lane/16)
// B 32x16: lane holds col N=lane%16; same K mapping as A
// C/D f32 16x16: element r -> M = r + half*8, N = lane%16
// ---------------------------------------------------------------------------
__device__ __forceinline__ bf16x16 load_a_frag(const __bf16* row_ptr, int half) {
  bf16x8 lo = *reinterpret_cast<const bf16x8*>(row_ptr + half * 8);
  bf16x8 hi = *reinterpret_cast<const bf16x8*>(row_ptr + half * 8 + 16);
  bf16x16 r;
#pragma unroll
  for (int i = 0; i < 8; ++i) { r[i] = lo[i]; r[i + 8] = hi[i]; }
  return r;
}

__device__ __forceinline__ bf16x16 load_b_frag(const __bf16* row_ptr, int half) {
  bf16x8 lo = *reinterpret_cast<const bf16x8*>(row_ptr + half * 8);
  bf16x8 hi = *reinterpret_cast<const bf16x8*>(row_ptr + half * 8 + 16);
  bf16x16 r;
#pragma unroll
  for (int i = 0; i < 8; ++i) { r[i] = lo[i]; r[i + 8] = hi[i]; }
  return r;
}

__device__ __forceinline__ f32x8 wmma_bf16(bf16x16 a, bf16x16 b, f32x8 c) {
  return __builtin_amdgcn_wmma_f32_16x16x32_bf16(
      /*neg_a=*/false, a, /*neg_b=*/false, b,
      /*c_mod=*/(short)0, c, /*reuse_a=*/false, /*reuse_b=*/false);
}

// ---------------------------------------------------------------------------
// Async global -> LDS staging (CDNA5 GLOBAL_LOAD_ASYNC_TO_LDS_B128, ASYNCcnt).
// LDS address = low 32 bits of the generic (flat) pointer.
// ---------------------------------------------------------------------------
__device__ __forceinline__ unsigned lds_off(const void* p) {
  return (unsigned)(unsigned long long)(size_t)p;
}

__device__ __forceinline__ void async_b128(unsigned dst_lds, const void* gsrc) {
  asm volatile("global_load_async_to_lds_b128 %0, %1, off"
               :: "v"(dst_lds), "v"((unsigned long long)gsrc)
               : "memory");
}

__device__ __forceinline__ void wait_async0() {
  asm volatile("s_wait_asynccnt 0x0" ::: "memory");
}

// ---------------------------------------------------------------------------
// Kernel 0a: fp32 -> bf16 conversion (grid-stride)
// ---------------------------------------------------------------------------
__global__ void cvt_f32_to_bf16(const float* __restrict__ src,
                                __bf16* __restrict__ dst, int n) {
  for (int i = blockIdx.x * blockDim.x + threadIdx.x; i < n;
       i += gridDim.x * blockDim.x) {
    dst[i] = (__bf16)src[i];
  }
}

// ---------------------------------------------------------------------------
// Kernel 0b: fp32 -> bf16 with [batch][rows][cols] -> [batch][cols][rows]
// transpose (one-time weight repack so GEMM B-tiles are K-contiguous).
// ---------------------------------------------------------------------------
__global__ void cvt_transpose_f32_to_bf16(const float* __restrict__ src,
                                          __bf16* __restrict__ dst,
                                          int rows, int cols, int total) {
  const int rc = rows * cols;
  for (int o = blockIdx.x * blockDim.x + threadIdx.x; o < total;
       o += gridDim.x * blockDim.x) {
    const int b = o / rc;
    const int rem = o - b * rc;
    const int c = rem / rows;       // output row (= src col)
    const int r = rem - c * rows;   // output col (= src row)
    dst[o] = (__bf16)src[(size_t)b * rc + (size_t)r * cols + c];
  }
}

// ---------------------------------------------------------------------------
// Kernel 1: QKV projection. grid = (M/64, H, 3), block = 256 (8 waves).
// Double-buffered async-to-LDS staging; 4 WMMAs per wave per K-step of 64.
// Weights arrive pre-transposed: Wt[h] is [D][E]. V stored as [B,H,D,S].
// ---------------------------------------------------------------------------
__global__ void __launch_bounds__(256)
qkv_gemm_kernel(const __bf16* __restrict__ xb,
                const __bf16* __restrict__ wqt,
                const __bf16* __restrict__ wkt,
                const __bf16* __restrict__ wvt,
                __bf16* __restrict__ qo,
                __bf16* __restrict__ ko,
                __bf16* __restrict__ vo) {
  __shared__ __bf16 As[2][64 * 64];   // x tile  [m][k]
  __shared__ __bf16 Bs[2][64 * 64];   // W tile  [n][k]

  const int h     = blockIdx.y;
  const int which = blockIdx.z;
  const __bf16* Wt = (which == 0) ? wqt : (which == 1) ? wkt : wvt;
  __bf16* O        = (which == 0) ? qo : (which == 1) ? ko : vo;
  Wt += (size_t)h * Dc * Ec;

  const int mbase = blockIdx.x * 64;
  const int t    = threadIdx.x;
  const int lane = t & 31, wid = t >> 5;
  const int l16  = lane & 15, half = lane >> 4;
  const int tr   = wid & 3;            // m sub-tile (0..3)
  const int tc0  = (wid >> 2) * 2;     // n sub-tiles tc0, tc0+1

  f32x8 acc0 = {}; f32x8 acc1 = {};

  // staging: 4096 bf16 per tile / 256 threads = 16 contiguous each
  const int sRow = (t * 16) / 64, sCol = (t * 16) % 64;
  const __bf16* aSrc = &xb[(size_t)(mbase + sRow) * Ec + sCol];
  const __bf16* bSrc = &Wt[(size_t)sRow * Ec + sCol];
  const unsigned aDst = lds_off(&As[0][sRow * 64 + sCol]);
  const unsigned bDst = lds_off(&Bs[0][sRow * 64 + sCol]);
  constexpr unsigned BUFB = 64 * 64 * sizeof(__bf16);  // 8 KB per buffer

  // prologue: stage tile 0 into buffer 0
  async_b128(aDst,      aSrc);
  async_b128(aDst + 16, aSrc + 8);
  async_b128(bDst,      bSrc);
  async_b128(bDst + 16, bSrc + 8);
  wait_async0();
  __syncthreads();

  int buf = 0;
  for (int k0 = 0; k0 < Ec; k0 += 64) {
    const int nbuf = buf ^ 1;
    if (k0 + 64 < Ec) {  // stage next tile while computing this one
      const unsigned bo = (unsigned)nbuf * BUFB;
      async_b128(aDst + bo,      aSrc + k0 + 64);
      async_b128(aDst + bo + 16, aSrc + k0 + 72);
      async_b128(bDst + bo,      bSrc + k0 + 64);
      async_b128(bDst + bo + 16, bSrc + k0 + 72);
      __builtin_prefetch(aSrc + k0 + 128, 0, 1);
      __builtin_prefetch(bSrc + k0 + 128, 0, 1);
    }

#pragma unroll
    for (int ks = 0; ks < 64; ks += 32) {
      bf16x16 a  = load_a_frag(&As[buf][(tr * 16 + l16) * 64 + ks], half);
      bf16x16 b0 = load_b_frag(&Bs[buf][(tc0 * 16 + l16) * 64 + ks], half);
      bf16x16 b1 = load_b_frag(&Bs[buf][((tc0 + 1) * 16 + l16) * 64 + ks], half);
      acc0 = wmma_bf16(a, b0, acc0);
      acc1 = wmma_bf16(a, b1, acc1);
    }

    wait_async0();
    __syncthreads();
    buf = nbuf;
  }

#pragma unroll
  for (int r = 0; r < 8; ++r) {
    const int m = mbase + tr * 16 + r + half * 8;
    const int b = m >> 11, s = m & (Sc - 1);
    const int d0 = tc0 * 16 + l16, d1 = (tc0 + 1) * 16 + l16;
    if (which == 2) {
      // V transposed: [B,H,D,S]
      const size_t base = (size_t)(b * Hc + h) * Dc * Sc;
      O[base + (size_t)d0 * Sc + s] = (__bf16)acc0[r];
      O[base + (size_t)d1 * Sc + s] = (__bf16)acc1[r];
    } else {
      // Q/K: [B,H,S,D]
      const size_t rowoff = ((size_t)(b * Hc + h) * Sc + s) * Dc;
      O[rowoff + d0] = (__bf16)acc0[r];
      O[rowoff + d1] = (__bf16)acc1[r];
    }
  }
}

// ---------------------------------------------------------------------------
// Kernel 2: causal flash attention. grid = (S/64, B*H), block = 128 (4 waves).
// Each wave owns 16 query rows; K/V staged async + double-buffered in 32-key
// blocks. V arrives pre-transposed [B,H,D,S].
// ---------------------------------------------------------------------------
__global__ void __launch_bounds__(128)
flash_attn_kernel(const __bf16* __restrict__ q,
                  const __bf16* __restrict__ k,
                  const __bf16* __restrict__ vT,
                  __bf16* __restrict__ attn) {
  __shared__ __bf16 Ks[2][32 * 64];     // [key][d]
  __shared__ __bf16 Vt[2][64 * 32];     // [d][key]
  __shared__ __bf16 Ps[4][16 * 32];     // per-wave P tile [qrow][key]

  const int bh = blockIdx.y;
  const int b = bh >> 4, h = bh & (Hc - 1);
  const int qb0 = blockIdx.x * 64;
  const int t = threadIdx.x;
  const int lane = t & 31, wid = t >> 5;
  const int l16 = lane & 15, half = lane >> 4;
  const int qw = qb0 + wid * 16;               // this wave's first query row
  const size_t headoff = (size_t)bh * Sc * Dc;

  // Q fragments for this wave (16 rows x 64 d)
  const __bf16* qrow = q + headoff + (size_t)(qw + l16) * Dc;
  bf16x16 aq0 = load_a_frag(qrow, half);
  bf16x16 aq1 = load_a_frag(qrow + 32, half);

  f32x8 O0 = {}, O1 = {}, O2 = {}, O3 = {};
  float mrow[8], lrow[8];
#pragma unroll
  for (int r = 0; r < 8; ++r) { mrow[r] = -__builtin_inff(); lrow[r] = 0.f; }

  // staging: 2048 elems per matrix / 128 threads = 16 each
  const int kRow = (t * 16) >> 6, kCol = (t * 16) & 63;   // [key][d]
  const int vRow = (t * 16) >> 5, vCol = (t * 16) & 31;   // [d][key]
  const __bf16* kSrc = k + headoff + (size_t)kRow * Dc + kCol;
  const __bf16* vSrc = vT + headoff + (size_t)vRow * Sc + vCol;
  const unsigned kDst = lds_off(&Ks[0][kRow * 64 + kCol]);
  const unsigned vDst = lds_off(&Vt[0][vRow * 32 + vCol]);
  constexpr unsigned BUFB = 32 * 64 * sizeof(__bf16);  // 4 KB per buffer

  // prologue: stage key block 0 into buffer 0
  async_b128(kDst,      kSrc);
  async_b128(kDst + 16, kSrc + 8);
  async_b128(vDst,      vSrc);
  async_b128(vDst + 16, vSrc + 8);
  wait_async0();
  __syncthreads();

  int buf = 0;
  const int nT = (qb0 + 63) / 32 + 1;
  for (int tkb = 0; tkb < nT; ++tkb) {
    const int kb = tkb * 32;
    if (tkb + 1 < nT) {  // stage next key block while computing this one
      const unsigned bo = (unsigned)(buf ^ 1) * BUFB;
      const int kn = kb + 32;
      async_b128(kDst + bo,      kSrc + (size_t)kn * Dc);
      async_b128(kDst + bo + 16, kSrc + (size_t)kn * Dc + 8);
      async_b128(vDst + bo,      vSrc + kn);
      async_b128(vDst + bo + 16, vSrc + kn + 8);
    }

    if (kb <= qw + 15) {  // wave-uniform causal skip
      // --- scores: two 16x16 tiles over 32 keys, contraction over D=64 ---
      f32x8 sc0 = {}, sc1 = {};
      {
        const __bf16* kr0 = &Ks[buf][l16 * 64];
        sc0 = wmma_bf16(aq0, load_b_frag(kr0, half), sc0);
        sc0 = wmma_bf16(aq1, load_b_frag(kr0 + 32, half), sc0);
        const __bf16* kr1 = &Ks[buf][(16 + l16) * 64];
        sc1 = wmma_bf16(aq0, load_b_frag(kr1, half), sc1);
        sc1 = wmma_bf16(aq1, load_b_frag(kr1 + 32, half), sc1);
      }
      // --- online softmax (rows live across 16 lanes of one half) ---
      const float scale = 0.125f;  // D^-0.5
      float p0[8], p1[8];
#pragma unroll
      for (int r = 0; r < 8; ++r) {
        const int qr = qw + r + half * 8;
        float s0 = sc0[r] * scale; if (kb + l16 > qr)      s0 = -__builtin_inff();
        float s1 = sc1[r] * scale; if (kb + 16 + l16 > qr) s1 = -__builtin_inff();
        float tm = fmaxf(s0, s1);
        for (int off = 1; off < 16; off <<= 1) tm = fmaxf(tm, __shfl_xor(tm, off, 32));
        const float mnew = fmaxf(mrow[r], tm);
        const float resc = __expf(mrow[r] - mnew);
        mrow[r] = mnew;
        const float e0 = __expf(s0 - mnew), e1 = __expf(s1 - mnew);
        float ts = e0 + e1;
        for (int off = 1; off < 16; off <<= 1) ts += __shfl_xor(ts, off, 32);
        lrow[r] = lrow[r] * resc + ts;
        O0[r] *= resc; O1[r] *= resc; O2[r] *= resc; O3[r] *= resc;
        p0[r] = e0; p1[r] = e1;
      }
      // --- C-layout -> A-layout transpose of P via per-wave LDS tile ---
#pragma unroll
      for (int r = 0; r < 8; ++r) {
        Ps[wid][(r + half * 8) * 32 + l16]      = (__bf16)p0[r];
        Ps[wid][(r + half * 8) * 32 + 16 + l16] = (__bf16)p1[r];
      }
      asm volatile("s_wait_dscnt 0x0" ::: "memory");
      bf16x16 ap = load_a_frag(&Ps[wid][l16 * 32], half);
      // --- O += P @ V  (four 16-wide d blocks) ---
      O0 = wmma_bf16(ap, load_b_frag(&Vt[buf][(0  + l16) * 32], half), O0);
      O1 = wmma_bf16(ap, load_b_frag(&Vt[buf][(16 + l16) * 32], half), O1);
      O2 = wmma_bf16(ap, load_b_frag(&Vt[buf][(32 + l16) * 32], half), O2);
      O3 = wmma_bf16(ap, load_b_frag(&Vt[buf][(48 + l16) * 32], half), O3);
    }

    wait_async0();
    __syncthreads();
    buf ^= 1;
  }

  // normalize and store heads-concatenated bf16 [B, S, H*D]
#pragma unroll
  for (int r = 0; r < 8; ++r) {
    const float inv = 1.f / lrow[r];
    const int s = qw + r + half * 8;
    const size_t off = ((size_t)(b * Sc + s)) * Ec + h * Dc;
    attn[off + 0  + l16] = (__bf16)(O0[r] * inv);
    attn[off + 16 + l16] = (__bf16)(O1[r] * inv);
    attn[off + 32 + l16] = (__bf16)(O2[r] * inv);
    attn[off + 48 + l16] = (__bf16)(O3[r] * inv);
  }
}

// ---------------------------------------------------------------------------
// Kernel 3: output projection + bias, fp32 out. grid = (M/64, E/64), block 256.
// Wo arrives pre-transposed: Wot is [N][K]. Same async double-buffered scheme.
// ---------------------------------------------------------------------------
__global__ void __launch_bounds__(256)
outproj_kernel(const __bf16* __restrict__ A,
               const __bf16* __restrict__ Wot,
               const float* __restrict__ bo,
               float* __restrict__ out) {
  __shared__ __bf16 As[2][64 * 64];
  __shared__ __bf16 Bs[2][64 * 64];

  const int mbase = blockIdx.x * 64;
  const int nbase = blockIdx.y * 64;
  const int t = threadIdx.x;
  const int lane = t & 31, wid = t >> 5;
  const int l16 = lane & 15, half = lane >> 4;
  const int tr = wid & 3, tc0 = (wid >> 2) * 2;

  f32x8 acc0 = {}; f32x8 acc1 = {};
  const int sRow = (t * 16) / 64, sCol = (t * 16) % 64;
  const __bf16* aSrc = &A[(size_t)(mbase + sRow) * Ec + sCol];
  const __bf16* bSrc = &Wot[(size_t)(nbase + sRow) * Ec + sCol];
  const unsigned aDst = lds_off(&As[0][sRow * 64 + sCol]);
  const unsigned bDst = lds_off(&Bs[0][sRow * 64 + sCol]);
  constexpr unsigned BUFB = 64 * 64 * sizeof(__bf16);

  async_b128(aDst,      aSrc);
  async_b128(aDst + 16, aSrc + 8);
  async_b128(bDst,      bSrc);
  async_b128(bDst + 16, bSrc + 8);
  wait_async0();
  __syncthreads();

  int buf = 0;
  for (int k0 = 0; k0 < Ec; k0 += 64) {
    const int nbuf = buf ^ 1;
    if (k0 + 64 < Ec) {
      const unsigned boff = (unsigned)nbuf * BUFB;
      async_b128(aDst + boff,      aSrc + k0 + 64);
      async_b128(aDst + boff + 16, aSrc + k0 + 72);
      async_b128(bDst + boff,      bSrc + k0 + 64);
      async_b128(bDst + boff + 16, bSrc + k0 + 72);
      __builtin_prefetch(aSrc + k0 + 128, 0, 1);
      __builtin_prefetch(bSrc + k0 + 128, 0, 1);
    }

#pragma unroll
    for (int ks = 0; ks < 64; ks += 32) {
      bf16x16 a  = load_a_frag(&As[buf][(tr * 16 + l16) * 64 + ks], half);
      bf16x16 b0 = load_b_frag(&Bs[buf][(tc0 * 16 + l16) * 64 + ks], half);
      bf16x16 b1 = load_b_frag(&Bs[buf][((tc0 + 1) * 16 + l16) * 64 + ks], half);
      acc0 = wmma_bf16(a, b0, acc0);
      acc1 = wmma_bf16(a, b1, acc1);
    }

    wait_async0();
    __syncthreads();
    buf = nbuf;
  }

#pragma unroll
  for (int r = 0; r < 8; ++r) {
    const int m = mbase + tr * 16 + r + half * 8;
    const int n0 = nbase + tc0 * 16 + l16;
    const int n1 = nbase + (tc0 + 1) * 16 + l16;
    out[(size_t)m * Ec + n0] = acc0[r] + bo[n0];
    out[(size_t)m * Ec + n1] = acc1[r] + bo[n1];
  }
}

// ---------------------------------------------------------------------------
// Host-side launch
// ---------------------------------------------------------------------------
extern "C" void kernel_launch(void* const* d_in, const int* in_sizes, int n_in,
                              void* d_out, int out_size, void* d_ws, size_t ws_size,
                              hipStream_t stream) {
  const float* x  = (const float*)d_in[0];
  const float* Wq = (const float*)d_in[1];
  const float* Wk = (const float*)d_in[2];
  const float* Wv = (const float*)d_in[3];
  const float* Wo = (const float*)d_in[4];
  const float* bo = (const float*)d_in[5];

  char* ws = (char*)d_ws;
  const size_t sz_x    = (size_t)Bc * Sc * Ec * sizeof(__bf16);        // 16 MB
  const size_t sz_w    = (size_t)Hc * Ec * Dc * sizeof(__bf16);        //  2 MB
  const size_t sz_head = (size_t)Bc * Hc * Sc * Dc * sizeof(__bf16);   // 16 MB

  __bf16* xb   = (__bf16*)(ws);                  size_t off = sz_x;
  __bf16* wqt  = (__bf16*)(ws + off);            off += sz_w;
  __bf16* wkt  = (__bf16*)(ws + off);            off += sz_w;
  __bf16* wvt  = (__bf16*)(ws + off);            off += sz_w;
  __bf16* wot  = (__bf16*)(ws + off);            off += sz_w;
  __bf16* qbuf = (__bf16*)(ws + off);            off += sz_head;
  __bf16* kbuf = (__bf16*)(ws + off);            off += sz_head;
  __bf16* vbuf = (__bf16*)(ws + off);            off += sz_head;  // [B,H,D,S]
  __bf16* attn = (__bf16*)(ws + off);            off += sz_x;

  // 1) fp32 -> bf16 (x) and fp32 -> bf16 + transpose (weights, one-time)
  cvt_f32_to_bf16<<<2048, 256, 0, stream>>>(x, xb, Bc * Sc * Ec);
  cvt_transpose_f32_to_bf16<<<512, 256, 0, stream>>>(Wq, wqt, Ec, Dc, Hc * Ec * Dc);
  cvt_transpose_f32_to_bf16<<<512, 256, 0, stream>>>(Wk, wkt, Ec, Dc, Hc * Ec * Dc);
  cvt_transpose_f32_to_bf16<<<512, 256, 0, stream>>>(Wv, wvt, Ec, Dc, Hc * Ec * Dc);
  cvt_transpose_f32_to_bf16<<<512, 256, 0, stream>>>(Wo, wot, Ec, Ec, Ec * Ec);

  // 2) QKV projections (V stored transposed [B,H,D,S])
  qkv_gemm_kernel<<<dim3(Mc / 64, Hc, 3), 256, 0, stream>>>(
      xb, wqt, wkt, wvt, qbuf, kbuf, vbuf);

  // 3) causal flash attention
  flash_attn_kernel<<<dim3(Sc / 64, Bc * Hc), 128, 0, stream>>>(
      qbuf, kbuf, vbuf, attn);

  // 4) output projection + bias
  outproj_kernel<<<dim3(Mc / 64, Ec / 64), 256, 0, stream>>>(
      attn, wot, bo, (float*)d_out);
}
